// MultiHeadAttention_65996467470601
// MI455X (gfx1250) — compile-verified, matrix operand order checked by reading, not run on hardware
//
#include <hip/hip_runtime.h>

// ---------------------------------------------------------------------------
// MHA with Gaussian prior, bf16 WMMA pipeline for gfx1250 (CDNA5, wave32).
// B=4, S=2048, F=512, H=4, DK=DV=128.
// ---------------------------------------------------------------------------

typedef __attribute__((ext_vector_type(16))) __bf16 bf16x16;
typedef __attribute__((ext_vector_type(8)))  __bf16 bf16x8;
typedef __attribute__((ext_vector_type(8)))  float  f32x8;
typedef __attribute__((ext_vector_type(4)))  float  f32x4;

#define B_  4
#define S_  2048
#define F_  512
#define H_  4
#define DK_ 128
#define DV_ 128

// Native conversions: gfx1250 has hardware f32<->bf16 converts; let clang
// lower these (v_cvt_pk_bf16_f32 / shift) instead of manual bit-twiddling.
__device__ __forceinline__ __bf16 f2bf(float f)  { return static_cast<__bf16>(f); }
__device__ __forceinline__ float  bf2f(__bf16 x) { return static_cast<float>(x); }

__device__ __forceinline__ f32x8 wmma_bf16(bf16x16 a, bf16x16 b, f32x8 c) {
  // 8 args: (neg_a, A, neg_b, B, c_mod, C, reuse_a, reuse_b)
  return __builtin_amdgcn_wmma_f32_16x16x32_bf16(false, a, false, b, (short)0, c,
                                                 false, false);
}

// A fragment (16xK32, 16-bit) from row-major bf16 source; same pattern serves
// the B operand when the source holds B^T row-major (row = output column).
// Lane l: row = l&15, kh = l>>4; elements k = kbase+kh*8..+7 and kbase+16+kh*8..+7.
__device__ __forceinline__ bf16x16 load_frag_bf16(const __bf16* src, int ld,
                                                  int row, int kbase, int kh) {
  const bf16x8* p0 = reinterpret_cast<const bf16x8*>(src + (size_t)row * ld + kbase + kh * 8);
  const bf16x8* p1 = reinterpret_cast<const bf16x8*>(src + (size_t)row * ld + kbase + 16 + kh * 8);
  bf16x8 lo = *p0, hi = *p1;
  bf16x16 r;
#pragma unroll
  for (int i = 0; i < 8; ++i) { r[i] = lo[i]; r[i + 8] = hi[i]; }
  return r;
}

// Same fragment but built from an fp32 row-major source (converted to bf16).
__device__ __forceinline__ bf16x16 load_frag_f32(const float* src, int ld,
                                                 int row, int kbase, int kh) {
  const f32x4* p = reinterpret_cast<const f32x4*>(src + (size_t)row * ld + kbase + kh * 8);
  f32x4 a0 = p[0], a1 = p[1];
  const f32x4* q = reinterpret_cast<const f32x4*>(src + (size_t)row * ld + kbase + 16 + kh * 8);
  f32x4 b0 = q[0], b1 = q[1];
  bf16x16 r;
#pragma unroll
  for (int i = 0; i < 4; ++i) {
    r[i]      = f2bf(a0[i]);
    r[i + 4]  = f2bf(a1[i]);
    r[i + 8]  = f2bf(b0[i]);
    r[i + 12] = f2bf(b1[i]);
  }
  return r;
}

// ---------------------------------------------------------------------------
// Kernel 1: W[512x512] fp32 -> transposed bf16  (Wt[n][k] = bf16(W[k][n]))
// ---------------------------------------------------------------------------
__global__ __launch_bounds__(256) void wconv_kernel(const float* __restrict__ W,
                                                    __bf16* __restrict__ Wt) {
  int idx = blockIdx.x * 256 + threadIdx.x;   // idx = k*512 + n  (coalesced read)
  int k = idx >> 9, n = idx & 511;
  Wt[(size_t)n * F_ + k] = f2bf(W[idx]);
}

// ---------------------------------------------------------------------------
// Kernel 2: projection  Y = X @ W  via WMMA.
// One wave computes a 16x64 output tile (4 column tiles share the A fragment).
// VLAYOUT==0 : out[b][h][s][d]  (Q, K)
// VLAYOUT==1 : out[b][h][d][s]  (V transposed, so P@V B-fragments are linear)
// ---------------------------------------------------------------------------
template <int VLAYOUT>
__global__ __launch_bounds__(256) void proj_kernel(const float* __restrict__ X,
                                                   const __bf16* __restrict__ Wt,
                                                   __bf16* __restrict__ out) {
  const int tid  = threadIdx.x;
  const int lane = tid & 31, wv = tid >> 5;
  const int kh = lane >> 4, r = lane & 15;
  const int wg = blockIdx.x * 8 + wv;         // 4096 waves total
  const int rowTile = wg >> 3;                // 0..511  (B*S/16)
  const int cg = wg & 7;                      // 0..7    (512/64 col groups)

  const f32x8 zero = {0.f, 0.f, 0.f, 0.f, 0.f, 0.f, 0.f, 0.f};
  f32x8 acc[4];
#pragma unroll
  for (int t = 0; t < 4; ++t) acc[t] = zero;

  for (int kb = 0; kb < F_; kb += 32) {
    bf16x16 a = load_frag_f32(X, F_, rowTile * 16 + r, kb, kh);
#pragma unroll
    for (int t = 0; t < 4; ++t) {
      bf16x16 b = load_frag_bf16(Wt, F_, (cg * 4 + t) * 16 + r, kb, kh);
      acc[t] = wmma_bf16(a, b, acc[t]);
    }
  }

#pragma unroll
  for (int t = 0; t < 4; ++t) {
    const int n = (cg * 4 + t) * 16 + r;      // output feature
    const int h = n >> 7, d = n & 127;
#pragma unroll
    for (int v = 0; v < 8; ++v) {
      const int rowg = rowTile * 16 + v + 8 * kh;   // token index over B*S
      const int bb = rowg >> 11, s = rowg & (S_ - 1);
      size_t idx;
      if (VLAYOUT) idx = (size_t)((bb * H_ + h) * DV_ + d) * S_ + s;
      else         idx = (size_t)((bb * H_ + h) * S_ + s) * DK_ + d;
      out[idx] = f2bf(acc[t][v]);
    }
  }
}

// ---------------------------------------------------------------------------
// Kernel 3: attention. One block = (b, h, 16-query stripe), 8 waves.
// LDS: 16 x 2048 bf16 score/probability stripe (64 KB dynamic).
// ---------------------------------------------------------------------------
__global__ __launch_bounds__(256) void attn_kernel(const __bf16* __restrict__ Qb,
                                                   const __bf16* __restrict__ Kb,
                                                   const __bf16* __restrict__ Vt,
                                                   const unsigned char* __restrict__ mask,
                                                   __bf16* __restrict__ ctx) {
  extern __shared__ __bf16 P[];               // [16][2048]
  const int tid  = threadIdx.x;
  const int lane = tid & 31, wv = tid >> 5;
  const int kh = lane >> 4, r = lane & 15;
  const int qt = blockIdx.x & 127;
  const int h  = (blockIdx.x >> 7) & 3;
  const int bb = blockIdx.x >> 9;

  const __bf16* Qbase = Qb + ((size_t)(bb * H_ + h) * S_ + qt * 16) * DK_;
  const __bf16* Kbase = Kb + (size_t)(bb * H_ + h) * S_ * DK_;
  const __bf16* Vbase = Vt + (size_t)(bb * H_ + h) * DV_ * S_;

  const float sigma[4] = {5.f, 10.f, 20.f, 40.f};
  const float inv2s2 = 1.f / (2.f * sigma[h] * sigma[h]);
  const float scale = 0.08838834764831845f;   // 1/sqrt(128)

  // Preload Q fragments (shared across all key tiles).
  bf16x16 qf[4];
#pragma unroll
  for (int c = 0; c < 4; ++c) qf[c] = load_frag_bf16(Qbase, DK_, r, c * 32, kh);

  // ---- Phase 1: scores = QK^T/sqrt(d) + causal Gaussian bias, -1e9 mask ----
  for (int kt = wv; kt < S_ / 16; kt += 8) {
    const f32x8 zero = {0.f, 0.f, 0.f, 0.f, 0.f, 0.f, 0.f, 0.f};
    f32x8 acc = zero;
#pragma unroll
    for (int c = 0; c < 4; ++c) {
      // B^T row n (key index) is just row-major K.
      bf16x16 bfr = load_frag_bf16(Kbase + (size_t)kt * 16 * DK_, DK_, r, c * 32, kh);
      acc = wmma_bf16(qf[c], bfr, acc);
    }
    const int j = kt * 16 + r;                // key index (C frag: col = lane&15)
#pragma unroll
    for (int v = 0; v < 8; ++v) {
      const int il = v + 8 * kh;              // local query row
      const int i  = qt * 16 + il;            // global query index
      float sc = acc[v] * scale;
      float diff = (float)(j - i);
      if (i >= j) sc += __expf(-diff * diff * inv2s2);
      if (mask[(size_t)bb * S_ * S_ + (size_t)i * S_ + j]) sc = -1e9f;
      P[il * S_ + j] = f2bf(sc);
    }
  }
  __syncthreads();

  // ---- Phase 2: row softmax over 2048 keys (16 lanes per row) ----
  {
    const int row = tid >> 4, c = tid & 15;
    float m = -3.402823466e38f;
    for (int j = c; j < S_; j += 16) m = fmaxf(m, bf2f(P[row * S_ + j]));
#pragma unroll
    for (int o = 8; o; o >>= 1) m = fmaxf(m, __shfl_xor(m, o, 16));
    float sum = 0.f;
    for (int j = c; j < S_; j += 16) {
      float e = __expf(bf2f(P[row * S_ + j]) - m);
      P[row * S_ + j] = f2bf(e);
      sum += e;
    }
#pragma unroll
    for (int o = 8; o; o >>= 1) sum += __shfl_xor(sum, o, 16);
    const float inv = 1.f / sum;
    for (int j = c; j < S_; j += 16)
      P[row * S_ + j] = f2bf(bf2f(P[row * S_ + j]) * inv);
  }
  __syncthreads();

  // ---- Phase 3: context = P @ V.  Wave wv owns DV columns [wv*16, wv*16+16) ----
  {
    const int n = wv * 16 + r;                // output dv column (B-frag row of V^T)
    const f32x8 zero = {0.f, 0.f, 0.f, 0.f, 0.f, 0.f, 0.f, 0.f};
    f32x8 acc = zero;
#pragma unroll 2
    for (int kc = 0; kc < S_; kc += 32) {
      bf16x16 a   = load_frag_bf16(P, S_, r, kc, kh);        // ds_load_b128 x2
      bf16x16 bfr = load_frag_bf16(Vbase, S_, n, kc, kh);    // V stored [d][s]
      acc = wmma_bf16(a, bfr, acc);
    }
#pragma unroll
    for (int v = 0; v < 8; ++v) {
      const int il = v + 8 * kh;
      const int s  = qt * 16 + il;
      size_t idx = ((size_t)(bb * S_ + s)) * (H_ * DV_) + h * DV_ + n;
      ctx[idx] = f2bf(acc[v]);
    }
  }
}

// ---------------------------------------------------------------------------
// Kernel 4: out = LayerNorm(ctx @ Wfc + residual).  One block = 16 token rows.
// ---------------------------------------------------------------------------
__global__ __launch_bounds__(256) void out_kernel(const __bf16* __restrict__ ctx,
                                                  const __bf16* __restrict__ Wfct,
                                                  const float* __restrict__ resid,
                                                  float* __restrict__ out) {
  __shared__ float Y[16][F_];
  __shared__ float red[16][16];
  const int tid  = threadIdx.x;
  const int lane = tid & 31, wv = tid >> 5;
  const int kh = lane >> 4, r = lane & 15;
  const int rowTile = blockIdx.x;             // 0..511

  const f32x8 zero = {0.f, 0.f, 0.f, 0.f, 0.f, 0.f, 0.f, 0.f};
  f32x8 acc[4];
#pragma unroll
  for (int t = 0; t < 4; ++t) acc[t] = zero;

  for (int kb = 0; kb < F_; kb += 32) {
    bf16x16 a = load_frag_bf16(ctx + (size_t)rowTile * 16 * F_, F_, r, kb, kh);
#pragma unroll
    for (int t = 0; t < 4; ++t) {
      bf16x16 b = load_frag_bf16(Wfct, F_, (wv * 4 + t) * 16 + r, kb, kh);
      acc[t] = wmma_bf16(a, b, acc[t]);
    }
  }

#pragma unroll
  for (int t = 0; t < 4; ++t) {
    const int n = (wv * 4 + t) * 16 + r;
#pragma unroll
    for (int v = 0; v < 8; ++v) {
      const int il = v + 8 * kh;
      Y[il][n] = acc[t][v] + resid[((size_t)rowTile * 16 + il) * F_ + n];
    }
  }
  __syncthreads();

  // LayerNorm per row (16 lanes per row).
  const int row = tid >> 4, c = tid & 15;
  float s = 0.f;
  for (int j = c; j < F_; j += 16) s += Y[row][j];
  red[row][c] = s;
  __syncthreads();
  float mu = 0.f;
#pragma unroll
  for (int t = 0; t < 16; ++t) mu += red[row][t];
  mu *= (1.f / F_);
  __syncthreads();
  float vs = 0.f;
  for (int j = c; j < F_; j += 16) { float d = Y[row][j] - mu; vs += d * d; }
  red[row][c] = vs;
  __syncthreads();
  float var = 0.f;
#pragma unroll
  for (int t = 0; t < 16; ++t) var += red[row][t];
  var *= (1.f / F_);
  const float rinv = rsqrtf(var + 1e-5f);
  for (int j = c; j < F_; j += 16)
    out[((size_t)rowTile * 16 + row) * F_ + j] = (Y[row][j] - mu) * rinv;
}

// ---------------------------------------------------------------------------
extern "C" void kernel_launch(void* const* d_in, const int* in_sizes, int n_in,
                              void* d_out, int out_size, void* d_ws, size_t ws_size,
                              hipStream_t stream) {
  (void)in_sizes; (void)n_in; (void)out_size; (void)ws_size;
  const float* inQ = (const float*)d_in[0];
  const float* inK = (const float*)d_in[1];
  const float* inV = (const float*)d_in[2];
  const unsigned char* mask = (const unsigned char*)d_in[3];
  const float* Wq  = (const float*)d_in[4];
  const float* Wk  = (const float*)d_in[5];
  const float* Wv  = (const float*)d_in[6];
  const float* Wfc = (const float*)d_in[7];

  char* ws = (char*)d_ws;
  const size_t WSZ = (size_t)F_ * F_ * 2;                 // 512 KB per bf16 weight
  const size_t QSZ = (size_t)B_ * H_ * S_ * DK_ * 2;      // 8 MB per bf16 tensor
  __bf16* Wqt = (__bf16*)(ws + 0 * WSZ);
  __bf16* Wkt = (__bf16*)(ws + 1 * WSZ);
  __bf16* Wvt = (__bf16*)(ws + 2 * WSZ);
  __bf16* Wft = (__bf16*)(ws + 3 * WSZ);
  size_t off = 4 * WSZ;
  __bf16* Qb  = (__bf16*)(ws + off); off += QSZ;
  __bf16* Kb  = (__bf16*)(ws + off); off += QSZ;
  __bf16* Vtw = (__bf16*)(ws + off); off += QSZ;
  __bf16* Ctx = (__bf16*)(ws + off); off += QSZ;

  // 1) weights -> transposed bf16
  wconv_kernel<<<(F_ * F_) / 256, 256, 0, stream>>>(Wq,  Wqt);
  wconv_kernel<<<(F_ * F_) / 256, 256, 0, stream>>>(Wk,  Wkt);
  wconv_kernel<<<(F_ * F_) / 256, 256, 0, stream>>>(Wv,  Wvt);
  wconv_kernel<<<(F_ * F_) / 256, 256, 0, stream>>>(Wfc, Wft);

  // 2) QKV projections (V transposed to [B,H,D,S])
  proj_kernel<0><<<512, 256, 0, stream>>>(inQ, Wqt, Qb);
  proj_kernel<0><<<512, 256, 0, stream>>>(inK, Wkt, Kb);
  proj_kernel<1><<<512, 256, 0, stream>>>(inV, Wvt, Vtw);

  // 3) attention: B*H*(S/16) = 2048 blocks, 64 KB dynamic LDS score stripe
  attn_kernel<<<B_ * H_ * (S_ / 16), 256, 16 * S_ * 2, stream>>>(Qb, Kb, Vtw, mask, Ctx);

  // 4) output projection + residual + layernorm
  out_kernel<<<(B_ * S_) / 16, 256, 0, stream>>>(Ctx, Wft, inQ, (float*)d_out);
}
